// Refine_LSTM_23605140258927
// MI455X (gfx1250) — matrix operand order repaired and checked by reading, not compiled, not run on hardware
//
#include <hip/hip_runtime.h>

// ---------------------------------------------------------------------------
// Refine-LSTM (GRU-D style) persistent bf16-WMMA kernel for gfx1250 (MI455X).
//  - weights packed per launch into WMMA B-fragment order (bf16)
//  - bf16 mirrors of all GEMM A-operands, written once at production time
//    (A-fragment load = two 16B global_load_b128, no per-reuse conversion)
//  - one persistent kernel walks all 256 timesteps, software grid barriers
//  - GEMMs: v_wmma_f32_16x16x32_bf16, f32 accumulate, fused epilogues
// ---------------------------------------------------------------------------

typedef unsigned short u16;
typedef unsigned int   u32;

typedef __bf16 bf16;
typedef bf16  v16bf __attribute__((ext_vector_type(16)));
typedef u16   v16us __attribute__((ext_vector_type(16)));
typedef float v8f   __attribute__((ext_vector_type(8)));
typedef u32   v8u   __attribute__((ext_vector_type(8)));
typedef u32   v4u   __attribute__((ext_vector_type(4)));

// problem dims
constexpr int Td = 256, Bd = 512, Fd = 128, Hd = 512, NCc = 3;
constexpr size_t BFe = (size_t)Bd * Fd;   // 65536
constexpr size_t BHe = (size_t)Bd * Hd;   // 262144

// launch config (persistent)
constexpr int NBLK = 128, NTHR = 256;

// ---- workspace layout -----------------------------------------------------
// u16 region: packed bf16 weights (B-fragment order) + bf16 activation mirrors
constexpr size_t OFF_Wdgh = 0;                           // [H x F]
constexpr size_t OFF_Wx   = OFF_Wdgh + (size_t)Hd * Fd;  // [F x H]
constexpr size_t OFF_Wxz  = OFF_Wx   + (size_t)Fd * Hd;  // [F x F] (diag zeroed)
constexpr size_t OFF_Wbdg = OFF_Wxz  + (size_t)Fd * Fd;  // [F x F]
constexpr size_t OFF_Wbr  = OFF_Wbdg + (size_t)Fd * Fd;  // [F x F]
constexpr size_t WG_SZ    = (size_t)Hd * (Fd + Hd);      // K=640 concat
constexpr size_t OFF_Wg0  = OFF_Wbr  + (size_t)Fd * Fd;  // 4 gates i,f,o,r
constexpr size_t OFF_Wc   = OFF_Wg0 + 4 * WG_SZ;         // candidate
constexpr size_t OFF_Wout = OFF_Wc  + WG_SZ;             // [F x H] (yh|gh)
// bf16 activation mirrors (row-major [rows][K])
constexpr size_t OFF_Ahb  = OFF_Wout + (size_t)Fd * Hd;  // h      [B x H]
constexpr size_t OFF_Ahdb = OFF_Ahb  + BHe;              // h_d    [B x H]
constexpr size_t OFF_Arhb = OFF_Ahdb + BHe;              // r*h_d  [B x H]
constexpr size_t OFF_Aub  = OFF_Arhb + BHe;              // u_t    [B x F]
constexpr size_t OFF_Axcb = OFF_Aub  + BFe;              // x_c    [B x F]
constexpr size_t OFF_Agxb = OFF_Axcb + BFe;              // gamma_x[B x F]
constexpr size_t OFF_Adb  = OFF_Agxb + BFe;              // Delta_t[B x F]
constexpr size_t OFF_Amb  = OFF_Adb  + BFe;              // Mask_t [B x F]
constexpr size_t USH_TOTAL= OFF_Amb  + BFe;
constexpr size_t FLOAT_BASE = ((USH_TOTAL * sizeof(u16) + 255) / 256) * 256;

// fp32 region offsets (floats)
constexpr size_t F_H    = 0;
constexpr size_t F_C    = F_H + BHe;
constexpr size_t F_HD   = F_C + BHe;
constexpr size_t F_GI   = F_HD + BHe;        // 4*BHe: gates i,f,o,r
constexpr size_t F_XHAT = F_GI + 4 * BHe;
constexpr size_t F_XC   = F_XHAT + BFe;
constexpr size_t F_BETA = F_XC + BFe;
constexpr size_t F_RHO  = F_BETA + BFe;
constexpr size_t F_ZHB  = F_RHO + BFe;
constexpr size_t F_BDGH = F_ZHB + BFe;       // Hd
constexpr size_t F_BG   = F_BDGH + Hd;       // 4*Hd (fused u+h biases)
constexpr size_t F_BC   = F_BG + 4 * Hd;     // Hd
constexpr size_t F_BX   = F_BC + Hd;         // Fd each below
constexpr size_t F_BXZ  = F_BX + Fd;
constexpr size_t F_BBDG = F_BXZ + Fd;
constexpr size_t F_BBR  = F_BBDG + Fd;
constexpr size_t F_BDGX = F_BBR + Fd;
constexpr size_t F_DIAG = F_BDGX + Fd;
constexpr size_t F_BOUT = F_DIAG + Fd;
constexpr size_t F_BAR  = F_BOUT + Fd;       // 8 floats for barrier counters

// d_out sections: outs[0..T-2], xh[T], zh[T], uh[T] (all [B,F] fp32)
constexpr size_t OUT_XH = (size_t)(Td - 1) * BFe;
constexpr size_t OUT_ZH = OUT_XH + (size_t)Td * BFe;
constexpr size_t OUT_UH = OUT_ZH + (size_t)Td * BFe;

// ---------------------------------------------------------------------------
__device__ inline u16 f2bf(float f) {            // fp32 -> bf16 RNE
  u32 u = __float_as_uint(f);
  return (u16)((u + 0x7FFFu + ((u >> 16) & 1u)) >> 16);
}
__device__ inline float sigm(float x) { return 1.f / (1.f + __expf(-x)); }
__device__ inline v8f vzero() { v8f z = {0.f,0.f,0.f,0.f,0.f,0.f,0.f,0.f}; return z; }

// A fragment 16x32 bf16 per ISA layout from a row-major bf16 buffer:
// lanes 0-15 (row M=lane) hold K k0+0..7 & k0+16..23; lanes 16-31 hold
// K k0+8..15 & k0+24..31.  Two aligned 16B loads per lane.
__device__ inline v16bf load_a_bf(const u16* A, int ldk, int m0, int k0, int lane) {
  const int row = m0 + (lane & 15);
  const int kh  = (lane >> 4) & 1;
  const u16* p = A + (size_t)row * ldk + k0 + kh * 8;
  const v4u lo = *(const v4u*)p;          // K .. K+7
  const v4u hi = *(const v4u*)(p + 16);   // K+16 .. K+23
  v8u c = {lo[0], lo[1], lo[2], lo[3], hi[0], hi[1], hi[2], hi[3]};
  return __builtin_bit_cast(v16bf, c);
}

// B fragment: pre-packed so each lane's 16 bf16 are contiguous (32B load).
__device__ inline v16bf load_b_frag(const u16* Wp, int ktiles, int nt, int kt, int lane) {
  const v8u* p = (const v8u*)(Wp + (((size_t)nt * ktiles + kt) * 32 + lane) * 16);
  return __builtin_bit_cast(v16bf, *p);
}

__device__ inline v8f wmma_bf16(v16bf a, v16bf b, v8f c) {
  return __builtin_amdgcn_wmma_f32_16x16x32_bf16(false, a, false, b, (short)0, c,
                                                 false, false);
}

// C = A[B x K] * W^T, 1xNT n-blocking (A-fragment reuse)
template <int NT>
__device__ inline void gemm_acc(const u16* A, int ldk, const u16* Wp, int ktiles,
                                int mt, int nt0, int lane, v8f (&acc)[NT]) {
  for (int kt = 0; kt < ktiles; ++kt) {
    v16bf a = load_a_bf(A, ldk, mt * 16, kt * 32, lane);
    if (kt + 1 < ktiles)   // weights are L2-resident; keep them streaming
      __builtin_prefetch(Wp + (((size_t)nt0 * ktiles + kt + 1) * 32 + lane) * 16, 0, 3);
#pragma unroll
    for (int j = 0; j < NT; ++j) {
      v16bf b = load_b_frag(Wp, ktiles, nt0 + j, kt, lane);
      acc[j] = wmma_bf16(a, b, acc[j]);
    }
  }
}

// same, A = [A1 (K1=kt1*32 cols) | A2] concatenated along K
template <int NT>
__device__ inline void gemm_acc_cat(const u16* A1, int ldk1, int kt1,
                                    const u16* A2, int ldk2,
                                    const u16* Wp, int ktiles,
                                    int mt, int nt0, int lane, v8f (&acc)[NT]) {
  for (int kt = 0; kt < ktiles; ++kt) {
    const u16* A = (kt < kt1) ? A1 : A2;
    const int ldk = (kt < kt1) ? ldk1 : ldk2;
    const int kb  = (kt < kt1) ? kt * 32 : (kt - kt1) * 32;
    v16bf a = load_a_bf(A, ldk, mt * 16, kb, lane);
#pragma unroll
    for (int j = 0; j < NT; ++j) {
      v16bf b = load_b_frag(Wp, ktiles, nt0 + j, kt, lane);
      acc[j] = wmma_bf16(a, b, acc[j]);
    }
  }
}

// software grid barrier (persistent kernel; grid sized to stay resident)
__device__ inline void grid_barrier(u32* bar) {
  __syncthreads();
  if (threadIdx.x == 0) {
    __threadfence();
    volatile u32* vgen = bar + 1;
    u32 g = *vgen;
    if (atomicAdd(bar, 1u) == (u32)(NBLK - 1)) {
      atomicExch(bar, 0u);
      __threadfence();
      *vgen = g + 1u;
    } else {
      while (*vgen == g) __builtin_amdgcn_s_sleep(1);
    }
    __threadfence();
  }
  __syncthreads();
}

// ---------------------------------------------------------------------------
// prep: pack a weight [N][K] fp32 into bf16 B-fragment order.
// mode 0: plain, 1: zero diagonal, 2: concat [s1(K1)|s2], 3: out = [yh(3)|gh]
__global__ void pack_weight(u16* dst, const float* s1, const float* s2,
                            int N, int K1, int K, int mode) {
  const int KT = K >> 5;
  const size_t total = (size_t)N * K;
  const size_t stride = (size_t)gridDim.x * blockDim.x;
  for (size_t i = (size_t)blockIdx.x * blockDim.x + threadIdx.x; i < total; i += stride) {
    const int e    = (int)(i & 15);
    const int lane = (int)((i >> 4) & 31);
    const size_t tile = i >> 9;
    const int kt = (int)(tile % KT);
    const int nt = (int)(tile / KT);
    const int n = nt * 16 + (lane & 15);
    const int k = kt * 32 + ((lane >> 4) << 3) + (e & 7) + ((e >> 3) << 4);   // B layout: K = kh*8 + (e&7) + (e>>3)*16
    float v;
    if (mode == 0)      v = s1[(size_t)n * K + k];
    else if (mode == 1) v = (n == k) ? 0.f : s1[(size_t)n * K + k];
    else if (mode == 2) v = (k < K1) ? s1[(size_t)n * K1 + k]
                                     : s2[(size_t)n * (K - K1) + (k - K1)];
    else                v = (n < NCc) ? s1[(size_t)n * K + k]
                                      : s2[(size_t)(n - NCc) * K + k];
    dst[i] = f2bf(v);
  }
}

// prep: zero h/c/hb, fuse biases, diag(w_dg_x), bf16 of Delta[0]/Mask[0], bar
__global__ void init_misc(float* wsf, u16* wsu,
    const float* D0, const float* M0,
    const float* b_dgh, const float* b_x, const float* b_xz, const float* b_bdg,
    const float* b_br,  const float* b_ui, const float* b_hi, const float* b_uf,
    const float* b_hf,  const float* b_uc, const float* b_hc, const float* b_uo,
    const float* b_ho,  const float* b_ur, const float* b_hr, const float* b_yh,
    const float* b_gh,  const float* w_dgx, const float* b_dgx) {
  const size_t tid = (size_t)blockIdx.x * blockDim.x + threadIdx.x;
  const size_t stride = (size_t)gridDim.x * blockDim.x;
  for (size_t j = tid; j < BHe; j += stride) {
    wsf[F_H + j] = 0.f; wsf[F_C + j] = 0.f;
    wsu[OFF_Ahb + j] = 0;                        // bf16 +0.0
  }
  for (size_t j = tid; j < BFe; j += stride) {
    wsu[OFF_Adb + j] = f2bf(D0[j]);
    wsu[OFF_Amb + j] = f2bf(M0[j]);
  }
  for (size_t j = tid; j < (size_t)Hd; j += stride) {
    wsf[F_BDGH + j] = b_dgh[j];
    wsf[F_BG + 0 * Hd + j] = b_ui[j] + b_hi[j];
    wsf[F_BG + 1 * Hd + j] = b_uf[j] + b_hf[j];
    wsf[F_BG + 2 * Hd + j] = b_uo[j] + b_ho[j];
    wsf[F_BG + 3 * Hd + j] = b_ur[j] + b_hr[j];
    wsf[F_BC + j] = b_uc[j] + b_hc[j];
  }
  for (size_t j = tid; j < (size_t)Fd; j += stride) {
    wsf[F_BX + j] = b_x[j];  wsf[F_BXZ + j] = b_xz[j];
    wsf[F_BBDG + j] = b_bdg[j]; wsf[F_BBR + j] = b_br[j];
    wsf[F_BDGX + j] = b_dgx[j];
    wsf[F_DIAG + j] = w_dgx[j * Fd + j];
    wsf[F_BOUT + j] = (j < (size_t)NCc) ? b_yh[j] : b_gh[j - NCc];
  }
  if (tid < 8) ((u32*)(wsf + F_BAR))[tid] = 0u;
}

// ---------------------------------------------------------------------------
__global__ __launch_bounds__(NTHR)
void refine_lstm_persistent(const float* __restrict__ X, const float* __restrict__ Mask,
                            const float* __restrict__ Delta, u16* __restrict__ wsu,
                            float* __restrict__ wsf, float* __restrict__ out) {
  const int lane = threadIdx.x & 31;
  const int wave = (int)((blockIdx.x * blockDim.x + threadIdx.x) >> 5);
  const int nw   = (NBLK * NTHR) >> 5;
  const size_t tid  = (size_t)blockIdx.x * blockDim.x + threadIdx.x;
  const size_t tstr = (size_t)NBLK * NTHR;

  float* h    = wsf + F_H;    float* c    = wsf + F_C;
  float* hd   = wsf + F_HD;
  float* gate = wsf + F_GI;                       // [4][B*H] i,f,o,r
  float* xhat = wsf + F_XHAT; float* xc   = wsf + F_XC;
  float* beta = wsf + F_BETA; float* rho  = wsf + F_RHO;
  float* zhb  = wsf + F_ZHB;
  u16* hb  = wsu + OFF_Ahb;   u16* hdb  = wsu + OFF_Ahdb;
  u16* rhb = wsu + OFF_Arhb;  u16* ub   = wsu + OFF_Aub;
  u16* xcb = wsu + OFF_Axcb;  u16* gxb  = wsu + OFF_Agxb;
  u16* dtb = wsu + OFF_Adb;   u16* mtb  = wsu + OFF_Amb;
  const float* bdgh = wsf + F_BDGH; const float* bg = wsf + F_BG;
  const float* bc = wsf + F_BC;     const float* bx = wsf + F_BX;
  const float* bxz = wsf + F_BXZ;   const float* bbdg = wsf + F_BBDG;
  const float* bbr = wsf + F_BBR;   const float* bdgx = wsf + F_BDGX;
  const float* diag = wsf + F_DIAG; const float* bout = wsf + F_BOUT;
  u32* bar = (u32*)(wsf + F_BAR);

  for (int t = 0; t < Td; ++t) {
    const float* Xt = X + (size_t)t * BFe;
    const float* Mt = Mask + (size_t)t * BFe;
    const float* Dt = Delta + (size_t)t * BFe;

    // ---- S1: gamma_h -> h_d ; x_hat (+x_c) ; gamma_x ----------------------
    for (int g = wave; g < 256; g += nw) {           // [B,H] K=F
      const int mt = g >> 3, nt0 = (g & 7) << 2;
      v8f acc[4] = {vzero(), vzero(), vzero(), vzero()};
      gemm_acc<4>(dtb, Fd, wsu + OFF_Wdgh, Fd / 32, mt, nt0, lane, acc);
#pragma unroll
      for (int j = 0; j < 4; ++j)
#pragma unroll
        for (int r = 0; r < 8; ++r) {
          const int row = mt * 16 + ((lane >> 4) << 3) + r;
          const int col = (nt0 + j) * 16 + (lane & 15);
          const float gv = __expf(-fmaxf(acc[j][r] + bdgh[col], 0.f));
          const size_t idx = (size_t)row * Hd + col;
          const float hdv = gv * h[idx];
          hd[idx] = hdv;
          hdb[idx] = f2bf(hdv);
        }
    }
    for (int g = wave; g < 64; g += nw) {            // [B,F] K=H : x_hat
      const int mt = g >> 1, nt0 = (g & 1) << 2;
      v8f acc[4] = {vzero(), vzero(), vzero(), vzero()};
      gemm_acc<4>(hb, Hd, wsu + OFF_Wx, Hd / 32, mt, nt0, lane, acc);
#pragma unroll
      for (int j = 0; j < 4; ++j)
#pragma unroll
        for (int r = 0; r < 8; ++r) {
          const int row = mt * 16 + ((lane >> 4) << 3) + r;
          const int col = (nt0 + j) * 16 + (lane & 15);
          const float v = acc[j][r] + bx[col];
          const size_t idx = (size_t)row * Fd + col;
          xhat[idx] = v;
          out[OUT_XH + (size_t)t * BFe + idx] = v;
          const float m = Mt[idx];
          const float xcv = m * Xt[idx] + (1.f - m) * v;   // x_c
          xc[idx] = xcv;
          xcb[idx] = f2bf(xcv);
        }
    }
    for (size_t i = tid; i < BFe; i += tstr) {       // gamma_x (diag linear)
      const int f = (int)(i & (size_t)(Fd - 1));
      gxb[i] = f2bf(__expf(-fmaxf(Dt[i] * diag[f] + bdgx[f], 0.f)));
    }
    grid_barrier(bar);

    // ---- S2: z_hat, beta, rho (three FxF GEMMs) ---------------------------
    for (int g = wave; g < 64; g += nw) {            // z_hat = x_c @ Wxz'
      const int mt = g >> 1, nt0 = (g & 1) << 2;
      v8f acc[4] = {vzero(), vzero(), vzero(), vzero()};
      gemm_acc<4>(xcb, Fd, wsu + OFF_Wxz, 4, mt, nt0, lane, acc);
#pragma unroll
      for (int j = 0; j < 4; ++j)
#pragma unroll
        for (int r = 0; r < 8; ++r) {
          const int row = mt * 16 + ((lane >> 4) << 3) + r;
          const int col = (nt0 + j) * 16 + (lane & 15);
          const float v = acc[j][r] + bxz[col];
          const size_t idx = (size_t)row * Fd + col;
          zhb[idx] = v;
          out[OUT_ZH + (size_t)t * BFe + idx] = v;
        }
    }
    for (int g = wave; g < 64; g += nw) {            // beta = sig(gx @ Wbdg)
      const int mt = g >> 1, nt0 = (g & 1) << 2;
      v8f acc[4] = {vzero(), vzero(), vzero(), vzero()};
      gemm_acc<4>(gxb, Fd, wsu + OFF_Wbdg, 4, mt, nt0, lane, acc);
#pragma unroll
      for (int j = 0; j < 4; ++j)
#pragma unroll
        for (int r = 0; r < 8; ++r) {
          const int row = mt * 16 + ((lane >> 4) << 3) + r;
          const int col = (nt0 + j) * 16 + (lane & 15);
          beta[(size_t)row * Fd + col] = sigm(acc[j][r] + bbdg[col]);
        }
    }
    for (int g = wave; g < 64; g += nw) {            // rho = sig(m @ Wbr)
      const int mt = g >> 1, nt0 = (g & 1) << 2;
      v8f acc[4] = {vzero(), vzero(), vzero(), vzero()};
      gemm_acc<4>(mtb, Fd, wsu + OFF_Wbr, 4, mt, nt0, lane, acc);
#pragma unroll
      for (int j = 0; j < 4; ++j)
#pragma unroll
        for (int r = 0; r < 8; ++r) {
          const int row = mt * 16 + ((lane >> 4) << 3) + r;
          const int col = (nt0 + j) * 16 + (lane & 15);
          rho[(size_t)row * Fd + col] = sigm(acc[j][r] + bbr[col]);
        }
    }
    grid_barrier(bar);

    // ---- S3a: imputation mix -> u_t (elementwise) -------------------------
    for (size_t i = tid; i < BFe; i += tstr) {
      const float bb = beta[i], rr = rho[i];
      float uh = bb * zhb[i] + (1.f - bb) * xhat[i];
      uh = rr * uh + (1.f - rr) * xc[i];
      out[OUT_UH + (size_t)t * BFe + i] = uh;
      const float m = Mt[i];
      ub[i] = f2bf(m * Xt[i] + (1.f - m) * uh);
    }
    grid_barrier(bar);

    // ---- S3b: gate GEMMs (K = F+H = 640, concat A = [u | h_d]) ------------
    for (int gt = 0; gt < 4; ++gt) {
      const u16* Wg = wsu + OFF_Wg0 + (size_t)gt * WG_SZ;
      float* gb = gate + (size_t)gt * BHe;
      const float* bgk = bg + (size_t)gt * Hd;
      for (int g = wave; g < 256; g += nw) {
        const int mt = g >> 3, nt0 = (g & 7) << 2;
        v8f acc[4] = {vzero(), vzero(), vzero(), vzero()};
        gemm_acc_cat<4>(ub, Fd, 4, hdb, Hd, Wg, 20, mt, nt0, lane, acc);
#pragma unroll
        for (int j = 0; j < 4; ++j)
#pragma unroll
          for (int r = 0; r < 8; ++r) {
            const int row = mt * 16 + ((lane >> 4) << 3) + r;
            const int col = (nt0 + j) * 16 + (lane & 15);
            const size_t idx = (size_t)row * Hd + col;
            const float s = sigm(acc[j][r] + bgk[col]);
            gb[idx] = s;
            if (gt == 3) rhb[idx] = f2bf(s * hd[idx]);   // r_g * h_d
          }
      }
    }
    grid_barrier(bar);

    // ---- S4: candidate + cell update (NT=1 -> all 1024 waves busy) --------
    for (int g = wave; g < 1024; g += nw) {
      const int mt = g >> 5, nt = g & 31;
      v8f acc[1] = {vzero()};
      gemm_acc_cat<1>(ub, Fd, 4, rhb, Hd, wsu + OFF_Wc, 20, mt, nt, lane, acc);
#pragma unroll
      for (int r = 0; r < 8; ++r) {
        const int row = mt * 16 + ((lane >> 4) << 3) + r;
        const int col = nt * 16 + (lane & 15);
        const size_t idx = (size_t)row * Hd + col;
        const float ctv = tanhf(acc[0][r] + bc[col]);
        const float cn = gate[1 * BHe + idx] * c[idx] + gate[0 * BHe + idx] * ctv;
        const float hn = gate[2 * BHe + idx] * tanhf(cn);
        c[idx] = cn;
        h[idx] = hn;
        hb[idx] = f2bf(hn);
      }
    }
    grid_barrier(bar);

    // ---- S5: output GEMM + 3-way softmax; prep next-step bf16 inputs ------
    for (int g = wave; g < 256; g += nw) {
      const int mt = g >> 3, nt = g & 7;
      v8f acc[1] = {vzero()};
      gemm_acc<1>(hb, Hd, wsu + OFF_Wout, 16, mt, nt, lane, acc);
#pragma unroll
      for (int r = 0; r < 8; ++r) {
        const int row = mt * 16 + ((lane >> 4) << 3) + r;
        const int col = nt * 16 + (lane & 15);
        float v = acc[0][r] + bout[col];
        if (nt == 0) {                               // cols 0..2 = class logits
          const int base = lane & 16;
          const float c0 = __shfl(v, base + 0, 32);
          const float c1 = __shfl(v, base + 1, 32);
          const float c2 = __shfl(v, base + 2, 32);
          const float mx = fmaxf(c0, fmaxf(c1, c2));
          const float s = __expf(c0 - mx) + __expf(c1 - mx) + __expf(c2 - mx);
          if ((lane & 15) < NCc) v = __expf(v - mx) / s;
        }
        if (t < Td - 1) out[(size_t)t * BFe + (size_t)row * Fd + col] = v;
      }
    }
    if (t + 1 < Td) {                                // bf16 of next-step inputs
      const float* Dn = Delta + (size_t)(t + 1) * BFe;
      const float* Mn = Mask + (size_t)(t + 1) * BFe;
      for (size_t i = tid; i < BFe; i += tstr) {
        dtb[i] = f2bf(Dn[i]);
        mtb[i] = f2bf(Mn[i]);
      }
    }
    grid_barrier(bar);
  }
}

// ---------------------------------------------------------------------------
extern "C" void kernel_launch(void* const* d_in, const int* in_sizes, int n_in,
                              void* d_out, int out_size, void* d_ws, size_t ws_size,
                              hipStream_t stream) {
  (void)in_sizes; (void)n_in; (void)out_size; (void)ws_size;
  const float* X     = (const float*)d_in[0];
  const float* Mask  = (const float*)d_in[1];
  const float* Delta = (const float*)d_in[2];
  // params order: for each name i -> weight at 3+2i, bias at 3+2i+1
  auto W  = [&](int i) { return (const float*)d_in[3 + 2 * i]; };
  auto Bv = [&](int i) { return (const float*)d_in[3 + 2 * i + 1]; };
  // 0 w_dg_x 1 w_dg_h 2 w_x 3 w_xz 4 w_b_dg 5 w_br 6 w_ui 7 w_hi 8 w_uf
  // 9 w_hf 10 w_uc 11 w_hc 12 w_uo 13 w_ho 14 w_ur 15 w_hr 16 w_yh 17 w_gh
  u16*   wsu = (u16*)d_ws;
  float* wsf = (float*)((char*)d_ws + FLOAT_BASE);

  pack_weight<<<512, 256, 0, stream>>>(wsu + OFF_Wdgh, W(1),  nullptr, Hd, 0, Fd, 0);
  pack_weight<<<512, 256, 0, stream>>>(wsu + OFF_Wx,   W(2),  nullptr, Fd, 0, Hd, 0);
  pack_weight<<<512, 256, 0, stream>>>(wsu + OFF_Wxz,  W(3),  nullptr, Fd, 0, Fd, 1);
  pack_weight<<<512, 256, 0, stream>>>(wsu + OFF_Wbdg, W(4),  nullptr, Fd, 0, Fd, 0);
  pack_weight<<<512, 256, 0, stream>>>(wsu + OFF_Wbr,  W(5),  nullptr, Fd, 0, Fd, 0);
  pack_weight<<<512, 256, 0, stream>>>(wsu + OFF_Wg0 + 0 * WG_SZ, W(6),  W(7),  Hd, Fd, Fd + Hd, 2);
  pack_weight<<<512, 256, 0, stream>>>(wsu + OFF_Wg0 + 1 * WG_SZ, W(8),  W(9),  Hd, Fd, Fd + Hd, 2);
  pack_weight<<<512, 256, 0, stream>>>(wsu + OFF_Wg0 + 2 * WG_SZ, W(12), W(13), Hd, Fd, Fd + Hd, 2);
  pack_weight<<<512, 256, 0, stream>>>(wsu + OFF_Wg0 + 3 * WG_SZ, W(14), W(15), Hd, Fd, Fd + Hd, 2);
  pack_weight<<<512, 256, 0, stream>>>(wsu + OFF_Wc,   W(10), W(11), Hd, Fd, Fd + Hd, 2);
  pack_weight<<<512, 256, 0, stream>>>(wsu + OFF_Wout, W(16), W(17), Fd, 0, Hd, 3);

  init_misc<<<1024, 256, 0, stream>>>(wsf, wsu, Delta, Mask,
      Bv(1), Bv(2), Bv(3), Bv(4), Bv(5), Bv(6), Bv(7), Bv(8), Bv(9),
      Bv(10), Bv(11), Bv(12), Bv(13), Bv(14), Bv(15), Bv(16), Bv(17),
      W(0), Bv(0));

  refine_lstm_persistent<<<NBLK, NTHR, 0, stream>>>(X, Mask, Delta, wsu, wsf,
                                                    (float*)d_out);
}
